// TestNet_13735305412642
// MI455X (gfx1250) — matrix operand order repaired and checked by reading, not compiled, not run on hardware
//
#include <hip/hip_runtime.h>

typedef __attribute__((ext_vector_type(16))) _Float16 v16h;
typedef __attribute__((ext_vector_type(8)))  float    v8f;
typedef __attribute__((ext_vector_type(4)))  unsigned int u32x4;
typedef __attribute__((ext_vector_type(8)))  int      i32x8;
typedef __attribute__((ext_vector_type(4)))  int      i32x4;

#define Wdim 2048
#define Hdim 512
#define SROW 2052             // f16 state row stride (cols 0..2049 readable)
#define NT 4                  // 16-col W tiles per wave: 32 waves * 4 = 128 tiles
#define THREADS 1024

// ---- Tensor Data Mover: DMA one 2048-float global row into LDS (6-arg builtin form).
// D# group0: [1:0]=count=1, [63:32]=lds_addr, [120:64]=global_addr, [127:126]=type(2).
// D# group1: [17:16]=data_size(2=4B), [63:48]=tensor_dim0, [95:80]=tensor_dim1,
//            [127:112]=tile_dim0, [143:128]=tile_dim1, [191:160]=tensor_dim0_stride.
__device__ __forceinline__ void tdm_load_row(const float* gsrc, unsigned lds_byte) {
    const unsigned long long ga = (unsigned long long)(uintptr_t)gsrc;
    u32x4 g0;
    g0[0] = 1u;
    g0[1] = lds_byte;
    g0[2] = (unsigned)ga;
    g0[3] = (unsigned)((ga >> 32) & 0x01FFFFFFull) | (2u << 30);
    i32x8 g1;
    g1[0] = (int)(2u << 16);                 // data_size = 4 bytes
    g1[1] = (int)((unsigned)Wdim << 16);     // tensor_dim0 = 2048
    g1[2] = (int)(1u << 16);                 // tensor_dim1 = 1
    g1[3] = (int)((unsigned)Wdim << 16);     // tile_dim0 = 2048
    g1[4] = 1;                               // tile_dim1 = 1
    g1[5] = Wdim;                            // tensor_dim0_stride
    g1[6] = 0;
    g1[7] = 0;
    const i32x4 z4 = {0, 0, 0, 0};
    const i32x8 z8 = {0, 0, 0, 0, 0, 0, 0, 0};
    __builtin_amdgcn_tensor_load_to_lds(g0, g1, z4, z4, z8, 0);
}

__global__ __launch_bounds__(THREADS)
void recurrent_conv_wmma(const float* __restrict__ xg,
                         const float* __restrict__ w1g, const float* __restrict__ b1g,
                         const float* __restrict__ w2g, const float* __restrict__ b2g,
                         const float* __restrict__ wlg, const float* __restrict__ blg,
                         const float* __restrict__ wog, const float* __restrict__ bog,
                         float* __restrict__ out)
{
    __shared__ _Float16 st[10 * SROW];          // recurrent state rows (ch 0..9), f16
    __shared__ float    xs[2 * Wdim + 16];      // x slices ch 10,11 (f32, TDM target)

    const int tid  = threadIdx.x;
    const int lane = tid & 31;
    const int wave = tid >> 5;
    const int n    = lane & 15;     // N column within 16-wide tile
    const int hi   = lane >> 4;     // lane half: K-range (B) / M-range (C)
    const int b    = blockIdx.x;    // one persistent workgroup per batch element

    const float s_bo = bog[0];
    const size_t xrow1 = ((size_t)b*3 + 1) * Hdim * Wdim;   // x[b,1,:,:]
    const size_t xrow2 = ((size_t)b*3 + 2) * Hdim * Wdim;   // x[b,2,:,:]

    // Kick off DMA of x[b,1,1,:] and x[b,2,1,:] -> xs; overlaps the step-0 head.
    if (wave == 0) {
        tdm_load_row(xg + xrow1 + (size_t)1*Wdim, (unsigned)(uintptr_t)(void*)&xs[0]);
        tdm_load_row(xg + xrow2 + (size_t)1*Wdim, (unsigned)(uintptr_t)(void*)&xs[Wdim]);
    }

    // ---- WMMA A operands (weights wl, f16, contiguous K = 16*hi + h), built once.
    v16h a0, a1;
    {
        const int mA = lane & 15;
        #pragma unroll
        for (int h = 0; h < 16; ++h) {
            const int K0 = h + (hi << 4);
            float f0 = 0.f, f1 = 0.f;
            if (mA < 10) {
                f0 = wlg[mA*36 + K0];
                const int K1 = K0 + 32;
                if (K1 < 36) f1 = wlg[mA*36 + K1];
            }
            a0[h] = (_Float16)f0;
            a1[h] = (_Float16)f1;
        }
    }
    float lbias[8], lwo[8];
    #pragma unroll
    for (int r = 0; r < 8; ++r) {
        const int m = (hi << 3) + r;
        lbias[r] = (m < 10) ? blg[m] : 0.f;
        lwo[r]   = (m < 10) ? wog[m] : 0.f;
    }

    // ---- Zero state rows (pad columns 0 / 2047 / >=2048 must stay 0) + xs tail.
    for (int i = tid; i < 10*SROW; i += THREADS) st[i] = (_Float16)0.f;
    if (tid < 16) xs[2*Wdim + tid] = 0.f;
    __syncthreads();

    // ---- Step-0 head: relu(conv 3->5 VALID) -> relu(conv 5->10 SAME); first_in read
    // straight from global into registers (one-time), fused 1x1 output conv.
    {
        const size_t o0 = (((size_t)b*3 + 0)*Hdim + (Hdim-1)) * Wdim;  // x[b,0,-1,:]
        const size_t o1 = xrow1;                                       // x[b,1,0,:]
        const size_t o2 = xrow2;                                       // x[b,2,0,:]
        for (int w = 1 + tid; w <= Wdim-2; w += THREADS) {
            float f0[5], f1[5], f2[5];
            #pragma unroll
            for (int d = 0; d < 5; ++d) {
                const int idx = w - 2 + d;
                const bool ok = (idx >= 0) && (idx < Wdim);
                f0[d] = ok ? xg[o0 + idx] : 0.f;
                f1[d] = ok ? xg[o1 + idx] : 0.f;
                f2[d] = ok ? xg[o2 + idx] : 0.f;
            }
            float acc[10];
            #pragma unroll
            for (int co = 0; co < 10; ++co) acc[co] = b2g[co];
            #pragma unroll
            for (int dk = 0; dk < 3; ++dk) {
                const int u = w - 1 + dk;               // padded-h1 column
                if (u >= 1 && u <= Wdim-2) {
                    float h1v[5];
                    #pragma unroll
                    for (int ci = 0; ci < 5; ++ci) {
                        float a = b1g[ci];
                        #pragma unroll
                        for (int kk = 0; kk < 3; ++kk) {
                            a += w1g[(ci*3 + 0)*3 + kk] * f0[dk+kk];
                            a += w1g[(ci*3 + 1)*3 + kk] * f1[dk+kk];
                            a += w1g[(ci*3 + 2)*3 + kk] * f2[dk+kk];
                        }
                        h1v[ci] = fmaxf(a, 0.f);
                    }
                    #pragma unroll
                    for (int co = 0; co < 10; ++co)
                        #pragma unroll
                        for (int ci = 0; ci < 5; ++ci)
                            acc[co] += w2g[(co*5 + ci)*3 + dk] * h1v[ci];
                }
            }
            float ov = s_bo;
            #pragma unroll
            for (int co = 0; co < 10; ++co) {
                const float v = fmaxf(acc[co], 0.f);
                st[co*SROW + w] = (_Float16)v;
                ov += v * wog[co];
            }
            out[((size_t)b*Hdim + 0)*Wdim + w] = ov;
        }
        if (tid < 2) out[((size_t)b*Hdim + 0)*Wdim + (tid ? Wdim-1 : 0)] = s_bo;
    }
    if (wave == 0) __builtin_amdgcn_s_wait_tensorcnt(0);   // x step-1 rows landed
    __syncthreads();

    // ---- Recurrence: 511 steps. Phase A = LDS reads + WMMA; Phase B = writeback + DMA.
    for (int t = 1; t < Hdim; ++t) {
        v8f c[NT];
        #pragma unroll
        for (int i = 0; i < NT; ++i) {
            const int tile = wave + 32*i;               // 0..127
            const int base = tile*16 + n;
            // Two pre-adjusted bases fold the hi-lane K offset into ds immediates:
            // h%3 in {0,1}: +5*SROW+1 ; h%3 == 2: +6*SROW-2.
            const _Float16* pA = st + base + (hi ? (5*SROW + 1) : 0);
            const _Float16* pB = st + base + (hi ? (6*SROW - 2) : 0);
            v16h b0;
            #pragma unroll
            for (int h = 0; h < 14; ++h) {
                const int offL = (h/3)*SROW + (h%3);
                b0[h] = ((h % 3) == 2) ? pB[offL] : pA[offL];
            }
            {   // h=14,15: lo lanes K=14,15 (state ch4/ch5); hi lanes K=30,31 (x ch10)
                _Float16 v14, v15;
                if (hi) {
                    v14 = (_Float16)xs[base + 0];
                    v15 = (_Float16)xs[base + 1];
                } else {
                    v14 = st[4*SROW + 2 + base];
                    v15 = st[5*SROW + 0 + base];
                }
                b0[14] = v14; b0[15] = v15;
            }
            v16h b1v;
            #pragma unroll
            for (int h = 0; h < 16; ++h) b1v[h] = (_Float16)0.f;
            if (!hi) {                                   // K=32..35 (x ch10 k2, ch11 k0..2)
                b1v[0] = (_Float16)xs[base + 2];
                b1v[1] = (_Float16)xs[Wdim + base + 0];
                b1v[2] = (_Float16)xs[Wdim + base + 1];
                b1v[3] = (_Float16)xs[Wdim + base + 2];
            }
            v8f cc = {};
            cc = __builtin_amdgcn_wmma_f32_16x16x32_f16(false, a0, false, b0,
                                                        (short)0, cc, false, false);
            cc = __builtin_amdgcn_wmma_f32_16x16x32_f16(false, a1, false, b1v,
                                                        (short)0, cc, false, false);
            c[i] = cc;
        }
        __syncthreads();   // its s_wait_dscnt(0) also retires all Phase-A xs reads

        // Launch next step's x DMA immediately -> overlaps all of Phase B.
        if (wave == 0 && t < Hdim-1) {
            tdm_load_row(xg + xrow1 + (size_t)(t+1)*Wdim, (unsigned)(uintptr_t)(void*)&xs[0]);
            tdm_load_row(xg + xrow2 + (size_t)(t+1)*Wdim, (unsigned)(uintptr_t)(void*)&xs[Wdim]);
        }
        // Phase B: bias+ReLU, write new state row, fused 1x1 output conv.
        #pragma unroll
        for (int i = 0; i < NT; ++i) {
            const int tile = wave + 32*i;
            const int w = tile*16 + n + 1;              // padded-row column
            float s = 0.f;
            #pragma unroll
            for (int r = 0; r < 8; ++r) {
                const float v = fmaxf(c[i][r] + lbias[r], 0.f);
                s = fmaf(v, lwo[r], s);
                const int m = (hi << 3) + r;
                if (m < 10 && w <= Wdim-2) st[m*SROW + w] = (_Float16)v;
            }
            const float tot = s + __shfl_xor(s, 16, 32);
            if (!hi && w <= Wdim-2) out[((size_t)b*Hdim + t)*Wdim + w] = tot + s_bo;
        }
        if (tid < 2) out[((size_t)b*Hdim + t)*Wdim + (tid ? Wdim-1 : 0)] = s_bo;
        if (t < Hdim-2) {   // nudge t+2 rows toward L2: global_prefetch_b8 (256B/lane)
            if (wave == 1) __builtin_prefetch(xg + xrow1 + (size_t)(t+2)*Wdim + lane*64, 0, 3);
            if (wave == 2) __builtin_prefetch(xg + xrow2 + (size_t)(t+2)*Wdim + lane*64, 0, 3);
        }
        if (wave == 0) __builtin_amdgcn_s_wait_tensorcnt(0);
        __syncthreads();
    }
}

extern "C" void kernel_launch(void* const* d_in, const int* in_sizes, int n_in,
                              void* d_out, int out_size, void* d_ws, size_t ws_size,
                              hipStream_t stream) {
    (void)in_sizes; (void)n_in; (void)d_ws; (void)ws_size; (void)out_size;
    const float* x  = (const float*)d_in[0];
    const float* w1 = (const float*)d_in[1];
    const float* b1 = (const float*)d_in[2];
    const float* w2 = (const float*)d_in[3];
    const float* b2 = (const float*)d_in[4];
    const float* wl = (const float*)d_in[5];
    const float* bl = (const float*)d_in[6];
    const float* wo = (const float*)d_in[7];
    const float* bo = (const float*)d_in[8];
    float* out = (float*)d_out;
    recurrent_conv_wmma<<<dim3(8), dim3(THREADS), 0, stream>>>(
        x, w1, b1, w2, b2, wl, bl, wo, bo, out);
}
